// Qwen3AttentionSeparated_13426067768041
// MI455X (gfx1250) — matrix operand order repaired and checked by reading, not compiled, not run on hardware
//
#include <hip/hip_runtime.h>
#include <hip/hip_bf16.h>
#include <cstdint>
#include <math.h>

// ---------------------------------------------------------------------------
// Qwen3 attention block for MI455X (gfx1250, wave32, WMMA + TDM).
// All four matmuls (QKV, Q*K^T, P*V, O-proj) run on v_wmma_f32_16x16x32_bf16.
// Attention V tiles are streamed LDS-side by the Tensor Data Mover with
// double buffering (TENSORcnt), giving hardware OOB zero-fill on the tail.
// ---------------------------------------------------------------------------

typedef __attribute__((ext_vector_type(16))) __bf16        v16bf;
typedef __attribute__((ext_vector_type(8)))  float         v8f;
typedef __attribute__((ext_vector_type(8)))  unsigned int  v8u;
typedef __attribute__((ext_vector_type(4)))  unsigned int  v4u;
typedef __attribute__((ext_vector_type(8)))  int           v8i;
typedef __attribute__((ext_vector_type(4)))  int           v4i;
typedef unsigned short u16;
typedef unsigned int   u32;
typedef unsigned long long u64;

#define SEQ      2048
#define HIDDEN   2048
#define NUM_H    16
#define NUM_KV   8
#define HEAD_D   128
#define QKV_N    4096              // Q(2048) + K(1024) + V(1024)
#define ATT_SCALE 0.08838834764831845f   // 1/sqrt(128)

// ----- helpers --------------------------------------------------------------

__device__ __forceinline__ u16 f2bf(float f) {
    u32 u = __builtin_bit_cast(u32, f);
    u32 r = u + 0x7FFFu + ((u >> 16) & 1u);      // round-to-nearest-even
    return (u16)(r >> 16);
}

__device__ __forceinline__ v8f v8f_zero() {
    v8f z = {0.f, 0.f, 0.f, 0.f, 0.f, 0.f, 0.f, 0.f};
    return z;
}

// Build a 16x bf16 fragment from two contiguous 16-byte chunks.
__device__ __forceinline__ v16bf mk_frag(uint4 a, uint4 b) {
    v8u p;
    p[0] = a.x; p[1] = a.y; p[2] = a.z; p[3] = a.w;
    p[4] = b.x; p[5] = b.y; p[6] = b.z; p[7] = b.w;
    return __builtin_bit_cast(v16bf, p);
}

__device__ __forceinline__ v8f wmma_bf16(v16bf a, v16bf b, v8f c) {
    return __builtin_amdgcn_wmma_f32_16x16x32_bf16(
        false, a, false, b, (short)0, c, false, false);
}

// TDM: DMA a 2D tile (32 rows x 128 bf16, row stride 1024 elements) from
// global memory into LDS. Rows beyond valid_rows are hardware zero-filled.
// Descriptor layout per CDNA5 ISA section 8.3/8.4. This toolchain exposes the
// 6-arg builtin: (uint32x4 g0, int32x8 g1, int32x4, int32x4, int32x8, i32 cpol).
__device__ __forceinline__ void tdm_load_v(const u16* gsrc, u16* ldst,
                                           int valid_rows) {
    u32 lds_off = (u32)(size_t)ldst;            // flat LDS addr: low 32 bits
    u64 ga = (u64)(size_t)gsrc;
    v4u g0;
    g0[0] = 1u;                                  // count=1, user mode
    g0[1] = lds_off;                             // lds_addr
    g0[2] = (u32)ga;                             // global_addr[31:0]
    g0[3] = ((u32)(ga >> 32) & 0x01FFFFFFu)      // global_addr[56:32]
            | 0x80000000u;                       // type=2 ("image")
    v8i g1;
    g1[0] = (int)0x00010000u;                    // data_size=1 (2 bytes)
    g1[1] = (int)(128u << 16);                   // tensor_dim0 = 128
    g1[2] = (int)(((u32)valid_rows & 0xFFFFu) << 16);   // tensor_dim1 lo16
    g1[3] = (int)(((u32)valid_rows >> 16) | (128u << 16)); // dim1 hi | tile_dim0=128
    g1[4] = 32;                                  // tile_dim1 = 32 rows
    g1[5] = 1024;                                // tensor_dim0_stride = 1024 elems
    g1[6] = 0;
    g1[7] = 0;
    v4i z4 = {0, 0, 0, 0};                       // 2-D tensor: groups 2/3 unused
    v8i z8 = {0, 0, 0, 0, 0, 0, 0, 0};
    __builtin_amdgcn_tensor_load_to_lds(g0, g1, z4, z4, z8, 0);
}

// ----- kernel 1: RMSNorm of hidden states -> bf16 ---------------------------

__global__ void __launch_bounds__(256)
rmsnorm_kernel(const float* __restrict__ x, const float* __restrict__ w,
               u16* __restrict__ out) {
    __shared__ float red[8];
    __shared__ float rms_s;
    const int s = blockIdx.x;
    const float* row = x + (size_t)s * HIDDEN;
    float ss = 0.f;
    for (int i = threadIdx.x; i < HIDDEN; i += 256) {
        float v = row[i];
        ss += v * v;
    }
    #pragma unroll
    for (int off = 16; off; off >>= 1) ss += __shfl_xor(ss, off, 32);
    if ((threadIdx.x & 31) == 0) red[threadIdx.x >> 5] = ss;
    __syncthreads();
    if (threadIdx.x == 0) {
        float t = 0.f;
        #pragma unroll
        for (int i = 0; i < 8; ++i) t += red[i];
        rms_s = rsqrtf(t / (float)HIDDEN + 1e-6f);
    }
    __syncthreads();
    const float rms = rms_s;
    for (int i = threadIdx.x; i < HIDDEN; i += 256)
        out[(size_t)s * HIDDEN + i] = f2bf(row[i] * rms * w[i]);
}

// ----- kernel 2: f32 -> bf16 pair convert (weights) -------------------------

__global__ void __launch_bounds__(256)
cvt_pairs_kernel(const float* __restrict__ src, u32* __restrict__ dst, int npairs) {
    for (int i = blockIdx.x * blockDim.x + threadIdx.x; i < npairs;
         i += gridDim.x * blockDim.x) {
        float a = src[2 * (size_t)i];
        float b = src[2 * (size_t)i + 1];
        dst[i] = (u32)f2bf(a) | ((u32)f2bf(b) << 16);
    }
}

// ----- kernel 3: slice V section of qkv (f32) -> bf16 -----------------------

__global__ void __launch_bounds__(256)
extract_v_kernel(const float* __restrict__ qkv, u32* __restrict__ dst, int npairs) {
    for (int i = blockIdx.x * blockDim.x + threadIdx.x; i < npairs;
         i += gridDim.x * blockDim.x) {
        int s = i >> 9;            // 512 pairs per row (1024 floats of V)
        int r = i & 511;
        const float* p = qkv + (size_t)s * QKV_N + (NUM_H + NUM_KV) * HEAD_D + 2 * r;
        dst[i] = (u32)f2bf(p[0]) | ((u32)f2bf(p[1]) << 16);
    }
}

// ----- kernel 4: WMMA GEMM  C[MxN] f32 = A[MxK] bf16 @ B[NxK]^T bf16 --------
// One wave computes a 16(M) x 64(N) strip: 4 accumulators, A reused 4x.
// Fragment layouts used here (wave32; A and B share the same k order):
//   A 16x32: lane m = lane&15; lanes 0-15 hold K 0..15, lanes 16-31 K 16..31
//   B 32x16: lane n = lane&15; lanes 0-15 hold K 0..15, lanes 16-31 K 16..31
//   C 16x16: VGPR r = row r (lanes 0-15) / row r+8 (lanes 16-31), col = lane&15

__global__ void __launch_bounds__(256)
gemm_bf16_nt(const u16* __restrict__ A, const u16* __restrict__ B,
             float* __restrict__ C, int M, int N, int K) {
    const int lane   = threadIdx.x & 31;
    const int wave   = blockIdx.x * (blockDim.x >> 5) + (threadIdx.x >> 5);
    const int ntiles = N >> 6;
    const int m0 = (wave / ntiles) << 4;
    const int n0 = (wave % ntiles) << 6;
    const int l15  = lane & 15;
    const int half = lane >> 4;
    const int kbA  = half << 3;   // 0 or 8
    const int kbB  = half << 4;   // 0 or 16

    const u16* arow = A + (size_t)(m0 + l15) * K;
    const u16* b0   = B + (size_t)(n0 + l15) * K;

    v8f acc0 = v8f_zero(), acc1 = v8f_zero(), acc2 = v8f_zero(), acc3 = v8f_zero();

    for (int k = 0; k < K; k += 32) {
        v16bf af = mk_frag(*(const uint4*)(arow + k + kbB),
                           *(const uint4*)(arow + k + kbB + 8));
        const u16* bk = b0 + k + kbB;
        v16bf bf0 = mk_frag(*(const uint4*)(bk),                  *(const uint4*)(bk + 8));
        v16bf bf1 = mk_frag(*(const uint4*)(bk + 16 * (size_t)K), *(const uint4*)(bk + 16 * (size_t)K + 8));
        v16bf bf2 = mk_frag(*(const uint4*)(bk + 32 * (size_t)K), *(const uint4*)(bk + 32 * (size_t)K + 8));
        v16bf bf3 = mk_frag(*(const uint4*)(bk + 48 * (size_t)K), *(const uint4*)(bk + 48 * (size_t)K + 8));
        acc0 = wmma_bf16(af, bf0, acc0);
        acc1 = wmma_bf16(af, bf1, acc1);
        acc2 = wmma_bf16(af, bf2, acc2);
        acc3 = wmma_bf16(af, bf3, acc3);
    }

    float* crow = C + (size_t)(m0 + half * 8) * N + n0 + l15;
    #pragma unroll
    for (int r = 0; r < 8; ++r) {
        crow[(size_t)r * N +  0] = acc0[r];
        crow[(size_t)r * N + 16] = acc1[r];
        crow[(size_t)r * N + 32] = acc2[r];
        crow[(size_t)r * N + 48] = acc3[r];
    }
}

// ----- kernel 5: RoPE + per-head RMSNorm -> bf16 Q/K ------------------------
// One wave per (s, head-unit); units 0..15 = Q heads, 16..23 = K heads.
// Lane handles rotation pairs (j, j+64) for j = lane and lane+32.

__global__ void __launch_bounds__(32)
rope_norm_kernel(const int* __restrict__ positions, const float* __restrict__ qkv,
                 const float* __restrict__ qw, const float* __restrict__ kw,
                 u16* __restrict__ Qb, u16* __restrict__ Kb) {
    const int unit = blockIdx.x;
    const int s    = unit / 24;
    const int h    = unit % 24;
    const bool isq = h < NUM_H;
    const int head = isq ? h : h - NUM_H;
    const float* row = qkv + (size_t)s * QKV_N +
                       (isq ? head * HEAD_D : NUM_H * HEAD_D + head * HEAD_D);
    const float* w = isq ? qw : kw;
    const float pos = (float)positions[s];
    const int lane = threadIdx.x;

    float y1v[2], y2v[2];
    float ssq = 0.f;
    #pragma unroll
    for (int t = 0; t < 2; ++t) {
        int j = lane + 32 * t;                         // 0..63
        float inv_freq = powf(10000.0f, -(float)j / 64.0f);
        float ang = pos * inv_freq;
        float c = cosf(ang), sn = sinf(ang);
        float x1 = row[j], x2 = row[j + 64];
        float y1 = x1 * c - x2 * sn;
        float y2 = x2 * c + x1 * sn;
        y1v[t] = y1; y2v[t] = y2;
        ssq += y1 * y1 + y2 * y2;
    }
    #pragma unroll
    for (int off = 16; off; off >>= 1) ssq += __shfl_xor(ssq, off, 32);
    const float rms = rsqrtf(ssq / (float)HEAD_D + 1e-6f);

    u16* out = isq ? (Qb + ((size_t)s * NUM_H + head) * HEAD_D)
                   : (Kb + ((size_t)s * NUM_KV + head) * HEAD_D);
    #pragma unroll
    for (int t = 0; t < 2; ++t) {
        int j = lane + 32 * t;
        out[j]      = f2bf(y1v[t] * rms * w[j]);
        out[j + 64] = f2bf(y2v[t] * rms * w[j + 64]);
    }
}

// ----- kernel 6: flash-style causal attention (WMMA + TDM) ------------------
// One wave per (head, 16-row m-tile). Streams 32-key blocks with online
// softmax. V tiles are DMA'd into LDS by the Tensor Data Mover with double
// buffering; the TDM zero-fills rows past the sequence end (OOB reads).
// LDS per wave: P tile (16x32 bf16) + 2x V tile (32x128 bf16) = 17,408 B.

__global__ void __launch_bounds__(64)
attn_kernel(const u16* __restrict__ Qb, const u16* __restrict__ Kb,
            const u16* __restrict__ Vb, u16* __restrict__ Ob) {
    __shared__ u16 lds[2 * 8704];                     // 2 waves * (512 + 2*4096)
    const int lane = threadIdx.x & 31;
    const int wv   = threadIdx.x >> 5;
    u16* Pl = lds + wv * 8704;
    u16* Vbuf0 = Pl + 512;
    u16* Vbuf1 = Pl + 512 + 4096;

    const int unit = blockIdx.x * 2 + wv;             // 0..2047
    const int head = unit & 15;
    const int mt   = unit >> 4;
    const int kvh  = head >> 1;                       // GQA: 2 Q heads per KV head
    const int m0   = mt << 4;
    const int l15  = lane & 15;
    const int half = lane >> 4;
    const int kbA  = half << 3;
    const int kbB  = half << 4;
    const int rowg_base = m0 + half * 8;

    // Q fragments for this m-tile (A operand), loaded once.
    const u16* qrow = Qb + ((size_t)(m0 + l15) * NUM_H + head) * HEAD_D;
    v16bf qf[4];
    #pragma unroll
    for (int kk = 0; kk < 4; ++kk)
        qf[kk] = mk_frag(*(const uint4*)(qrow + kk * 32 + kbB),
                         *(const uint4*)(qrow + kk * 32 + kbB + 8));

    float rmax[8], rsum[8];
    v8f o[8];
    #pragma unroll
    for (int r = 0; r < 8; ++r) { rmax[r] = -__builtin_inff(); rsum[r] = 0.f; }
    #pragma unroll
    for (int j = 0; j < 8; ++j) o[j] = v8f_zero();

    const int nblocks = ((m0 + 15) >> 5) + 1;

    // Prologue: TDM prefetch of the first V tile.
    tdm_load_v(Vb + (size_t)kvh * HEAD_D, Vbuf0, SEQ);

    for (int nb = 0; nb < nblocks; ++nb) {
        const int n0 = nb << 5;

        // TDM prefetch of the next V tile into the other buffer.
        if (nb + 1 < nblocks) {
            const int n1 = (nb + 1) << 5;
            tdm_load_v(Vb + ((size_t)n1 * NUM_KV + kvh) * HEAD_D,
                       ((nb + 1) & 1) ? Vbuf1 : Vbuf0, SEQ - n1);
        }

        // Scores S = Q K^T for two 16x16 tiles (keys n0..n0+31).
        v8f sc[2] = { v8f_zero(), v8f_zero() };
        #pragma unroll
        for (int t = 0; t < 2; ++t) {
            int kidx = n0 + 16 * t + l15; if (kidx > SEQ - 1) kidx = SEQ - 1;
            const u16* krow = Kb + ((size_t)kidx * NUM_KV + kvh) * HEAD_D;
            #pragma unroll
            for (int kk = 0; kk < 4; ++kk) {
                v16bf kf = mk_frag(*(const uint4*)(krow + kk * 32 + kbB),
                                   *(const uint4*)(krow + kk * 32 + kbB + 8));
                sc[t] = wmma_bf16(qf[kk], kf, sc[t]);
            }
        }

        // Online softmax (rows live in 16-lane halves; reduce with xor<16).
        #pragma unroll
        for (int r = 0; r < 8; ++r) {
            const int rowg = rowg_base + r;
            float v0 = sc[0][r] * ATT_SCALE;
            float v1 = sc[1][r] * ATT_SCALE;
            if (n0 + l15      > rowg) v0 = -__builtin_inff();
            if (n0 + 16 + l15 > rowg) v1 = -__builtin_inff();
            float bm = fmaxf(v0, v1);
            #pragma unroll
            for (int off = 8; off; off >>= 1) bm = fmaxf(bm, __shfl_xor(bm, off, 32));
            float nm   = fmaxf(rmax[r], bm);
            float corr = expf(rmax[r] - nm);
            rmax[r] = nm;
            float p0 = expf(v0 - nm);
            float p1 = expf(v1 - nm);
            float ps = p0 + p1;
            #pragma unroll
            for (int off = 8; off; off >>= 1) ps += __shfl_xor(ps, off, 32);
            rsum[r] = rsum[r] * corr + ps;
            #pragma unroll
            for (int j = 0; j < 8; ++j) o[j][r] *= corr;
            // P tile to LDS, row-major 16x32, bf16.
            Pl[(r + half * 8) * 32 + l15]      = f2bf(p0);
            Pl[(r + half * 8) * 32 + 16 + l15] = f2bf(p1);
        }

        // Intra-wave LDS store->load ordering (DS is in-order per wave).
        asm volatile("s_wait_dscnt 0" ::: "memory");
        // Current V tile must be resident; the (nb+1) prefetch may stay in flight.
        if (nb + 1 < nblocks) __builtin_amdgcn_s_wait_tensorcnt(1);
        else                  __builtin_amdgcn_s_wait_tensorcnt(0);

        const u16* Vl = (nb & 1) ? Vbuf1 : Vbuf0;

        // P as A operand (16x32), V columns as B operands (32x16 each).
        v16bf pf = mk_frag(*(const uint4*)(Pl + l15 * 32 + kbB),
                           *(const uint4*)(Pl + l15 * 32 + kbB + 8));
        #pragma unroll
        for (int j = 0; j < 8; ++j) {
            v8u pw;
            #pragma unroll
            for (int w = 0; w < 8; ++w) {
                u16 lo = Vl[(kbB + 2 * w)     * HEAD_D + j * 16 + l15];
                u16 hi = Vl[(kbB + 2 * w + 1) * HEAD_D + j * 16 + l15];
                pw[w] = (u32)lo | ((u32)hi << 16);
            }
            v16bf vf = __builtin_bit_cast(v16bf, pw);
            o[j] = wmma_bf16(pf, vf, o[j]);
        }
    }

    // Epilogue: normalize by row sums and emit bf16 [S][NUM_H*HEAD_D].
    u16* orow = Ob + (size_t)rowg_base * (NUM_H * HEAD_D) + head * HEAD_D + l15;
    #pragma unroll
    for (int r = 0; r < 8; ++r) {
        float inv = 1.0f / rsum[r];
        #pragma unroll
        for (int j = 0; j < 8; ++j)
            orow[(size_t)r * (NUM_H * HEAD_D) + j * 16] = f2bf(o[j][r] * inv);
    }
}

// ----- launcher -------------------------------------------------------------

extern "C" void kernel_launch(void* const* d_in, const int* in_sizes, int n_in,
                              void* d_out, int out_size, void* d_ws, size_t ws_size,
                              hipStream_t stream) {
    (void)in_sizes; (void)n_in; (void)out_size; (void)ws_size;

    const int*   positions = (const int*)  d_in[0];
    const float* hidden    = (const float*)d_in[1];
    const float* lnw       = (const float*)d_in[2];
    const float* qkvw      = (const float*)d_in[3];
    const float* qnw       = (const float*)d_in[4];
    const float* knw       = (const float*)d_in[5];
    const float* ow        = (const float*)d_in[6];
    float*       out       = (float*)d_out;

    char* ws = (char*)d_ws;
    size_t off = 0;
    auto take = [&](size_t bytes) -> char* {
        char* p = ws + off;
        off += (bytes + 255) & ~(size_t)255;
        return p;
    };
    u16*   normA  = (u16*)  take((size_t)SEQ * HIDDEN * 2);          // 8 MB
    u16*   qkvWb  = (u16*)  take((size_t)QKV_N * HIDDEN * 2);        // 16 MB
    u16*   oWb    = (u16*)  take((size_t)HIDDEN * HIDDEN * 2);       // 8 MB
    float* qkvF   = (float*)take((size_t)SEQ * QKV_N * 4);           // 32 MB
    u16*   Qb     = (u16*)  take((size_t)SEQ * NUM_H * HEAD_D * 2);  // 8 MB
    u16*   Kb     = (u16*)  take((size_t)SEQ * NUM_KV * HEAD_D * 2); // 4 MB
    u16*   Vb     = (u16*)  take((size_t)SEQ * NUM_KV * HEAD_D * 2); // 4 MB
    u16*   AOb    = (u16*)  take((size_t)SEQ * NUM_H * HEAD_D * 2);  // 8 MB

    // 1. RMSNorm hidden -> bf16
    rmsnorm_kernel<<<SEQ, 256, 0, stream>>>(hidden, lnw, normA);
    // 2. Weights -> bf16
    cvt_pairs_kernel<<<2048, 256, 0, stream>>>(qkvw, (u32*)qkvWb, QKV_N * HIDDEN / 2);
    cvt_pairs_kernel<<<1024, 256, 0, stream>>>(ow,   (u32*)oWb,   HIDDEN * HIDDEN / 2);
    // 3. QKV GEMM: (2048x2048) @ (4096x2048)^T -> f32
    //    wave tiles = (2048/16)*(4096/64) = 8192, 8 waves/block -> 1024 blocks
    gemm_bf16_nt<<<1024, 256, 0, stream>>>(normA, qkvWb, qkvF, SEQ, QKV_N, HIDDEN);
    // 4. RoPE + q/k head RMSNorm -> bf16
    rope_norm_kernel<<<SEQ * 24, 32, 0, stream>>>(positions, qkvF, qnw, knw, Qb, Kb);
    // 5. V slice -> bf16
    extract_v_kernel<<<1024, 256, 0, stream>>>(qkvF, (u32*)Vb, SEQ * 512);
    // 6. Causal GQA attention -> bf16 (2048 wave-units, 2 waves/block)
    attn_kernel<<<1024, 64, 0, stream>>>(Qb, Kb, Vb, AOb);
    // 7. O projection: (2048x2048) @ (2048x2048)^T -> f32 out
    //    wave tiles = 128*32 = 4096 -> 512 blocks
    gemm_bf16_nt<<<512, 256, 0, stream>>>(AOb, oWb, out, SEQ, HIDDEN, HIDDEN);
}
